// AtariGRU_37726992728626
// MI455X (gfx1250) — compile-verified, matrix operand order checked by reading, not run on hardware
//
#include <hip/hip_runtime.h>

// ---------------------------------------------------------------------------
// GRU (Keras reset_after=True) scan for MI455X / gfx1250.
// B=256, T=256, F=512, U=1024.
//
//  * Pre-pass: transpose+convert [kernel(512x3072); recurrent(1024x3072)] into
//    bf16 W_t[3072][1536] in d_ws (9.4 MB, L2-resident for the whole scan).
//  * Scan: 16 persistent blocks of 1024 threads (32 wave32s), each owning 16
//    batch rows. The activation panel (x_t || masked h) lives in LDS as a
//    double-buffered bf16 [16][1536] panel (96 KB). Carried h state stays in
//    registers (each lane owns the same C-tile elements every step). Per step
//    each wave computes 2 column tiles of 16 units with chained
//    v_wmma_f32_16x16x32_bf16 (K=1536 for z/r; split K for the h-gate so r
//    multiplies only the recurrent part), applies gates, writes h to the
//    alternate panel + seq output. Exactly one workgroup barrier per step.
// ---------------------------------------------------------------------------

typedef __attribute__((ext_vector_type(16))) __bf16 v16bf;
typedef __attribute__((ext_vector_type(8)))  __bf16 v8bf;
typedef __attribute__((ext_vector_type(4)))  __bf16 v4bf;
typedef __attribute__((ext_vector_type(8)))  float  v8f;
typedef __attribute__((ext_vector_type(4)))  float  v4f;

#define B_   256
#define T_   256
#define F_   512
#define U_   1024
#define K_   1536            // F_ + U_
#define N3_  3072            // 3 * U_
#define ROWS 16              // batch rows per block (one WMMA M tile)
#define PAN  (ROWS * K_)     // bf16 elements per panel buffer (24576)

// ---------------------------------------------------------------------------
// Phase 0: W_t[n][k] = bf16( k < F ? kernel[k][n] : recurrent[k-F][n] )
// ---------------------------------------------------------------------------
__global__ void wt_convert(const float* __restrict__ kern,
                           const float* __restrict__ rker,
                           __bf16* __restrict__ wt) {
  int idx = blockIdx.x * blockDim.x + threadIdx.x;
  if (idx >= N3_ * K_) return;
  int n = idx % N3_;
  int k = idx / N3_;
  float v = (k < F_) ? kern[(size_t)k * N3_ + n]
                     : rker[(size_t)(k - F_) * N3_ + n];
  wt[(size_t)n * K_ + k] = (__bf16)v;
}

// ---------------------------------------------------------------------------
// B fragment (wave32 16x16x32 bf16 layout, CDNA5 ISA 7.12.2): lane = N because
// we read the transposed weights W_t (rows = N, contiguous K); two 8-element
// K runs per lane -> two 16B loads.
// ---------------------------------------------------------------------------
__device__ inline v16bf load_b_frag(const __bf16* __restrict__ rowp,
                                    int o1, int o2) {
  v8bf lo = *(const v8bf*)(rowp + o1);
  v8bf hi = *(const v8bf*)(rowp + o2);
  v16bf r;
#pragma unroll
  for (int j = 0; j < 8; ++j) {
    r[j]     = lo[j];
    r[j + 8] = hi[j];
  }
  return r;
}

__device__ inline float sigmoid_f(float v) {
  return 1.0f / (1.0f + __expf(-v));
}

__device__ inline v4bf cvt4(v4f v, float mk) {
  v4bf c;
#pragma unroll
  for (int j = 0; j < 4; ++j) c[j] = (__bf16)(v[j] * mk);
  return c;
}

// ---------------------------------------------------------------------------
// Persistent GRU scan. Grid = B/ROWS = 16 blocks, 1024 threads each.
// Dynamic LDS: bf16 panel[2][ROWS][K_]  (96 KB).
// ---------------------------------------------------------------------------
__launch_bounds__(1024, 1)
__global__ void gru_scan(const float* __restrict__ x,      // [B,T,F]
                         const int*   __restrict__ dones,  // [B,T]
                         const float* __restrict__ h0,     // [B,U]
                         const float* __restrict__ bias,   // [2,3U]
                         const __bf16* __restrict__ wt,    // [3U,K]
                         float* __restrict__ out_seq,      // [B,T,U]
                         float* __restrict__ out_hfin) {   // [B,U]
  extern __shared__ __align__(16) char smemraw[];
  __bf16* pan = (__bf16*)smemraw;          // [2][ROWS][K_], integer-indexed

  const int tid  = threadIdx.x;
  const int lane = tid & 31;
  const int wave = tid >> 5;               // 0..31
  const int ncol = lane & 15;              // N index inside tile
  const int half = lane >> 4;              // K-half selector
  const int b0   = blockIdx.x * ROWS;

  // carried hidden state: lane owns C-tile elems (M = e + 8*half, N = ncol)
  float hreg[2][8];
#pragma unroll
  for (int tile = 0; tile < 2; ++tile) {
    const int u0 = (wave + tile * 32) * 16;
#pragma unroll
    for (int e = 0; e < 8; ++e) {
      const int m = e + half * 8;
      hreg[tile][e] = h0[(size_t)(b0 + m) * U_ + u0 + ncol];
    }
  }

  // ---- prologue: stage bf16 x_0 and masked bf16 h0 into panel 0 ----------
  for (int i = tid; i < ROWS * (F_ / 4); i += blockDim.x) {
    const int m = i / (F_ / 4), f4 = (i % (F_ / 4)) * 4;
    v4f v = *(const v4f*)&x[((size_t)(b0 + m) * T_) * F_ + f4];
    *(v4bf*)&pan[m * K_ + f4] = cvt4(v, 1.0f);
  }
  for (int i = tid; i < ROWS * (U_ / 4); i += blockDim.x) {
    const int m = i / (U_ / 4), u4 = (i % (U_ / 4)) * 4;
    const float mk = dones[(size_t)(b0 + m) * T_] ? 0.0f : 1.0f;
    v4f v = *(const v4f*)&h0[(size_t)(b0 + m) * U_ + u4];
    *(v4bf*)&pan[m * K_ + F_ + u4] = cvt4(v, mk);
  }
  __syncthreads();

  for (int t = 0; t < T_; ++t) {
    const int pcur = (t & 1) * PAN;
    const int pnxt = PAN - pcur;

    // -- stage bf16 x_{t+1} into the other panel (compute only reads pcur,
    //    and only the h-region of pnxt is written by the update code) -------
    if (t + 1 < T_) {
      for (int i = tid; i < ROWS * (F_ / 4); i += blockDim.x) {
        const int m = i / (F_ / 4), f4 = (i % (F_ / 4)) * 4;
        v4f v = *(const v4f*)&x[((size_t)(b0 + m) * T_ + (t + 1)) * F_ + f4];
        *(v4bf*)&pan[pnxt + m * K_ + f4] = cvt4(v, 1.0f);
      }
    }

    const int abase = pcur + ncol * K_;    // this lane's A row (x_t || h)

#pragma unroll
    for (int tile = 0; tile < 2; ++tile) {
      const int u0 = (wave + tile * 32) * 16;
      const __bf16* bzr = wt + ((size_t)(0 * U_ + u0 + ncol)) * K_;
      const __bf16* brr = wt + ((size_t)(1 * U_ + u0 + ncol)) * K_;
      const __bf16* bhr = wt + ((size_t)(2 * U_ + u0 + ncol)) * K_;
      __builtin_prefetch(bzr, 0, 3);
      __builtin_prefetch(brr, 0, 3);
      __builtin_prefetch(bhr, 0, 3);

      v8f acc_z  = {0.f, 0.f, 0.f, 0.f, 0.f, 0.f, 0.f, 0.f};
      v8f acc_r  = {0.f, 0.f, 0.f, 0.f, 0.f, 0.f, 0.f, 0.f};
      v8f acc_xh = {0.f, 0.f, 0.f, 0.f, 0.f, 0.f, 0.f, 0.f};
      v8f acc_rh = {0.f, 0.f, 0.f, 0.f, 0.f, 0.f, 0.f, 0.f};

      // K steps 0..15 : x-region; h-gate accumulates xh.
      for (int ks = 0; ks < F_ / 32; ++ks) {
        const int k0 = ks * 32;
        const int o1 = k0 + half * 8;
        const int o2 = o1 + 16;
        v8bf alo = *(const v8bf*)&pan[abase + o1];
        v8bf ahi = *(const v8bf*)&pan[abase + o2];
        v16bf a;
#pragma unroll
        for (int j = 0; j < 8; ++j) { a[j] = alo[j]; a[j + 8] = ahi[j]; }
        v16bf bz = load_b_frag(bzr, o1, o2);
        v16bf br = load_b_frag(brr, o1, o2);
        v16bf bh = load_b_frag(bhr, o1, o2);
        acc_z  = __builtin_amdgcn_wmma_f32_16x16x32_bf16(false, a, false, bz,
                                                         (short)0, acc_z, false, false);
        acc_r  = __builtin_amdgcn_wmma_f32_16x16x32_bf16(false, a, false, br,
                                                         (short)0, acc_r, false, false);
        acc_xh = __builtin_amdgcn_wmma_f32_16x16x32_bf16(false, a, false, bh,
                                                         (short)0, acc_xh, false, false);
      }
      // K steps 16..47 : h-region (mask pre-applied); h-gate accumulates rh.
      for (int ks = F_ / 32; ks < K_ / 32; ++ks) {
        const int k0 = ks * 32;
        const int o1 = k0 + half * 8;
        const int o2 = o1 + 16;
        v8bf alo = *(const v8bf*)&pan[abase + o1];
        v8bf ahi = *(const v8bf*)&pan[abase + o2];
        v16bf a;
#pragma unroll
        for (int j = 0; j < 8; ++j) { a[j] = alo[j]; a[j + 8] = ahi[j]; }
        v16bf bz = load_b_frag(bzr, o1, o2);
        v16bf br = load_b_frag(brr, o1, o2);
        v16bf bh = load_b_frag(bhr, o1, o2);
        acc_z  = __builtin_amdgcn_wmma_f32_16x16x32_bf16(false, a, false, bz,
                                                         (short)0, acc_z, false, false);
        acc_r  = __builtin_amdgcn_wmma_f32_16x16x32_bf16(false, a, false, br,
                                                         (short)0, acc_r, false, false);
        acc_rh = __builtin_amdgcn_wmma_f32_16x16x32_bf16(false, a, false, bh,
                                                         (short)0, acc_rh, false, false);
      }

      // -- gate math + state update (C/D layout: elem e -> M = e + 8*half,
      //    N = ncol; one bias value per lane) --------------------------------
      const float bz2 = bias[u0 + ncol]           + bias[N3_ + u0 + ncol];
      const float br2 = bias[U_ + u0 + ncol]      + bias[N3_ + U_ + u0 + ncol];
      const float bxh = bias[2 * U_ + u0 + ncol];
      const float brh = bias[N3_ + 2 * U_ + u0 + ncol];
#pragma unroll
      for (int e = 0; e < 8; ++e) {
        const int m = e + half * 8;
        const size_t drow = (size_t)(b0 + m) * T_;
        const float mk_cur = dones[drow + t] ? 0.0f : 1.0f;
        const float z  = sigmoid_f(acc_z[e] + bz2);
        const float r  = sigmoid_f(acc_r[e] + br2);
        const float hh = tanhf(acc_xh[e] + bxh + r * (acc_rh[e] + brh));
        const float ho = hreg[tile][e] * mk_cur;
        const float hn = z * ho + (1.0f - z) * hh;
        hreg[tile][e] = hn;
        out_seq[(drow + t) * U_ + u0 + ncol] = hn;
        if (t + 1 < T_) {
          const float mk_nxt = dones[drow + t + 1] ? 0.0f : 1.0f;
          pan[pnxt + m * K_ + F_ + u0 + ncol] = (__bf16)(hn * mk_nxt);
        } else {
          out_hfin[(size_t)(b0 + m) * U_ + u0 + ncol] = hn;
        }
      }
    }
    __syncthreads();   // h(t+1) panel + x(t+1) staging complete
  }
}

// ---------------------------------------------------------------------------
extern "C" void kernel_launch(void* const* d_in, const int* in_sizes, int n_in,
                              void* d_out, int out_size, void* d_ws, size_t ws_size,
                              hipStream_t stream) {
  const float* x     = (const float*)d_in[0];
  const int*   dones = (const int*)d_in[1];
  const float* h0    = (const float*)d_in[2];
  const float* kern  = (const float*)d_in[3];
  const float* rker  = (const float*)d_in[4];
  const float* bias  = (const float*)d_in[5];
  float*  out = (float*)d_out;
  __bf16* wt  = (__bf16*)d_ws;       // 3072*1536*2 = 9.4 MB

  {
    const int total = N3_ * K_;
    const int thr   = 256;
    wt_convert<<<(total + thr - 1) / thr, thr, 0, stream>>>(kern, rker, wt);
  }

  const size_t smem = (size_t)2 * PAN * sizeof(__bf16);   // 96 KB
  (void)hipFuncSetAttribute(reinterpret_cast<const void*>(gru_scan),
                            hipFuncAttributeMaxDynamicSharedMemorySize,
                            (int)smem);
  gru_scan<<<dim3(B_ / ROWS), dim3(1024), smem, stream>>>(
      x, dones, h0, bias, wt, out, out + (size_t)B_ * T_ * U_);
}
